// BahdanauAttention_60653528154151
// MI455X (gfx1250) — matrix operand order repaired and checked by reading, not compiled, not run on hardware
//
#include <hip/hip_runtime.h>
#include <cstdint>
#include <cstddef>

#define B_ 16
#define T_ 4096
#define H_ 1024

// score_kernel: 128 t-rows per workgroup
#define ROWS_PER_WG 128
#define SMEM_A_BYTES (ROWS_PER_WG * H_ * 2)            // 256 KB bf16 swizzled keys
#define SMEM_QP_OFF  SMEM_A_BYTES                      // 4 KB q_proj row
#define SMEM_V_OFF   (SMEM_A_BYTES + H_ * 4)           // 4 KB v
#define SMEM_BYTES   (SMEM_A_BYTES + 2 * H_ * 4)       // 270336 B total

typedef __bf16 bf16_t;
typedef __attribute__((ext_vector_type(16))) __bf16 v16bf;
typedef __attribute__((ext_vector_type(2)))  __bf16 v2bf;
typedef __attribute__((ext_vector_type(8)))  float  v8f;

// ---- f32 -> bf16 (hardware V_CVT_PK_BF16_F32, probe-confirmed this toolchain)
__device__ __forceinline__ v2bf f2bf2(float lo, float hi) {
#if __has_builtin(__builtin_amdgcn_cvt_pk_bf16_f32)
    return __builtin_amdgcn_cvt_pk_bf16_f32(lo, hi);
#else
    v2bf r; r[0] = (bf16_t)lo; r[1] = (bf16_t)hi; return r;
#endif
}

// ---- tanh (hardware V_TANH_F32 transcendental if available) ----------------
__device__ __forceinline__ float fast_tanh(float x) {
#if __has_builtin(__builtin_amdgcn_tanhf)
    return __builtin_amdgcn_tanhf(x);
#elif __has_builtin(__builtin_amdgcn_tanh_f32)
    return __builtin_amdgcn_tanh_f32(x);
#else
    return tanhf(x);
#endif
}

// ---------------------------------------------------------------------------
// Kernel 1: q_proj[b,g] = sum_h query[b,h] * Wq[g,h]   (f32, tiny: 33 MFLOP)
// ---------------------------------------------------------------------------
__global__ __launch_bounds__(256)
void qproj_kernel(const float* __restrict__ query, const float* __restrict__ Wq,
                  float* __restrict__ qproj) {
    const int b = blockIdx.x;
    __shared__ float q[H_];
    for (int i = threadIdx.x; i < H_; i += 256) q[i] = query[(size_t)b * H_ + i];
    __syncthreads();
    for (int g = threadIdx.x; g < H_; g += 256) {
        const float4* wr = (const float4*)(Wq + (size_t)g * H_);
        float acc = 0.0f;
        for (int h4 = 0; h4 < H_ / 4; ++h4) {
            float4 w = wr[h4];
            int h = h4 * 4;
            acc += q[h] * w.x + q[h + 1] * w.y + q[h + 2] * w.z + q[h + 3] * w.w;
        }
        qproj[b * H_ + g] = acc;
    }
}

// ---------------------------------------------------------------------------
// Kernel 2: Wk (f32 row-major [g][h]) -> bf16, pre-swizzled per-lane into the
// WMMA B-matrix (32x16 K-major) layout.  B[k][n] = Wk[g = n][h = k].
// Lane = column n (lane&15); lanes 0-15 hold K=0..15, lanes 16-31 K=16..31.
// Tile (gt, kc): 32 lanes x 16 bf16 (32 bytes) each.
// ---------------------------------------------------------------------------
__global__ __launch_bounds__(256)
void wk_swizzle_kernel(const float* __restrict__ Wk, bf16_t* __restrict__ wsB) {
    const int tid  = threadIdx.x;
    const int wave = tid >> 5, lane = tid & 31;
    const int tile = blockIdx.x * 8 + wave;       // 0..2047 = gt*32 + kc
    const int gt = tile >> 5, kc = tile & 31;
    const int g = gt * 16 + (lane & 15);
    const int half = lane >> 4;
    const float* row = Wk + (size_t)g * H_ + kc * 32 + half * 16;
    union { v2bf p[8]; uint4 q[2]; } tmp;
#pragma unroll
    for (int i = 0; i < 8; ++i) tmp.p[i] = f2bf2(row[2 * i], row[2 * i + 1]);
    bf16_t* dst = wsB + ((size_t)tile * 32 + lane) * 16;
    *(uint4*)(dst)     = tmp.q[0];
    *(uint4*)(dst + 8) = tmp.q[1];
}

// ---------------------------------------------------------------------------
// Kernel 3 (hot): one workgroup = 128 t-rows of one batch.
// Phase 0: stream keys tile (coalesced f32), hw-cvt to bf16, store A-swizzled
//          into 256 KB LDS; stage q_proj row + v in LDS.
// Phase 1: wave w owns M-tile w (16 rows). All waves sweep the same B tiles
//          (WGP$ dedupes); gt unrolled x8 for 8 independent WMMA acc chains.
//          score[t] = sum_g tanh(qp[g]+kp[t,g])*v[g], reduced via shfl.
// ---------------------------------------------------------------------------
__global__ __launch_bounds__(256)
void score_kernel(const float* __restrict__ keys, const bf16_t* __restrict__ wsB,
                  const float* __restrict__ qproj, const float* __restrict__ vvec,
                  float* __restrict__ score) {
    extern __shared__ __align__(16) char smem[];
    float* qp_s = (float*)(smem + SMEM_QP_OFF);
    float* v_s  = (float*)(smem + SMEM_V_OFF);

    const int b    = blockIdx.y;                  // 0..15
    const int t0   = blockIdx.x * ROWS_PER_WG;    // 0..3968
    const int tid  = threadIdx.x;
    const int wave = tid >> 5;
    const int lane = tid & 31;
    const int half = lane >> 4;
    const int mrow = lane & 15;

    // ---- Phase 0: keys tile -> bf16, A-swizzled in LDS ----------------------
    // A tile (mt, kc) at byte offset (mt*32+kc)*1024; within tile lane*32 bytes
// holding K' = half*16..+15 (elems 0-15).
    for (int idx = tid; idx < ROWS_PER_WG * (H_ / 4); idx += 256) {
        const int row = idx >> 8;                 // 0..127
        const int k   = (idx & 255) << 2;         // 0..1020, step 4
        float4 w = *(const float4*)(keys + ((size_t)(b * T_ + t0 + row)) * H_ + k);
        union { v2bf p[2]; uint2 u; } st;
        st.p[0] = f2bf2(w.x, w.y);
        st.p[1] = f2bf2(w.z, w.w);
        const int kc = k >> 5, kk = k & 31;
        int h, elem;
        h = kk >> 4; elem = kk & 15;
        const int lane_d = h * 16 + (row & 15);
        const int mt = row >> 4;
        *(uint2*)(smem + (((mt << 5) + kc) << 10) + lane_d * 32 + elem * 2) = st.u;
    }
    for (int i = tid; i < H_; i += 256) {
        qp_s[i] = qproj[b * H_ + i];
        v_s[i]  = vvec[i];
    }
    __syncthreads();

    // ---- Phase 1: WMMA sweep ------------------------------------------------
    const v16bf* bsrc = (const v16bf*)wsB;
    const char*  abase = smem + ((wave << 5) << 10) + lane * 32;  // this wave's M-tile

    float partial[8];
#pragma unroll
    for (int r = 0; r < 8; ++r) partial[r] = 0.0f;

    for (int g8 = 0; g8 < 8; ++g8) {
        v8f acc[8] = {};
        for (int kc = 0; kc < 32; ++kc) {
            v16bf a = *(const v16bf*)(abase + (kc << 10));
#pragma unroll
            for (int j = 0; j < 8; ++j) {
                const int gt = g8 * 8 + j;
                v16bf bm = bsrc[(size_t)(gt * 32 + kc) * 32 + lane];
                acc[j] = __builtin_amdgcn_wmma_f32_16x16x32_bf16(
                    false, a, false, bm, (short)0, acc[j], false, false);
            }
        }
        // epilogue for these 8 g-tiles: C layout: VGPR r -> row r+8*half, col mrow
#pragma unroll
        for (int j = 0; j < 8; ++j) {
            const int g  = (g8 * 8 + j) * 16 + mrow;
            const float qp = qp_s[g];
            const float vg = v_s[g];
#pragma unroll
            for (int r = 0; r < 8; ++r)
                partial[r] += fast_tanh(qp + acc[j][r]) * vg;
        }
    }

    // Deterministic reduce over g (16 lanes per half), then store 16 scores.
#pragma unroll
    for (int m = 1; m < 16; m <<= 1)
#pragma unroll
        for (int r = 0; r < 8; ++r)
            partial[r] += __shfl_xor(partial[r], m, 16);

    if (mrow == 0) {
#pragma unroll
        for (int r = 0; r < 8; ++r)
            score[b * T_ + t0 + wave * 16 + 8 * half + r] = partial[r];
    }
}

// ---------------------------------------------------------------------------
// Kernel 4: softmax over T per batch (deterministic tree reductions)
// ---------------------------------------------------------------------------
__global__ __launch_bounds__(256)
void softmax_kernel(const float* __restrict__ score, float* __restrict__ attn_out,
                    float* __restrict__ ws_attn) {
    const int b = blockIdx.x, tid = threadIdx.x;
    __shared__ float red[256];
    float m = -1e30f;
    for (int t = tid; t < T_; t += 256) m = fmaxf(m, score[b * T_ + t]);
    red[tid] = m; __syncthreads();
    for (int s = 128; s > 0; s >>= 1) {
        if (tid < s) red[tid] = fmaxf(red[tid], red[tid + s]);
        __syncthreads();
    }
    const float mx = red[0]; __syncthreads();
    float sum = 0.0f;
    for (int t = tid; t < T_; t += 256) sum += expf(score[b * T_ + t] - mx);
    red[tid] = sum; __syncthreads();
    for (int s = 128; s > 0; s >>= 1) {
        if (tid < s) red[tid] += red[tid + s];
        __syncthreads();
    }
    const float inv = 1.0f / red[0];
    for (int t = tid; t < T_; t += 256) {
        float a = expf(score[b * T_ + t] - mx) * inv;
        attn_out[b * T_ + t] = a;
        ws_attn[b * T_ + t]  = a;
    }
}

// ---------------------------------------------------------------------------
// Kernel 5: context[b,h] = sum_t attn[b,t] * keys[b,t,h]   (streams keys once)
// ---------------------------------------------------------------------------
__global__ __launch_bounds__(256)
void context_kernel(const float* __restrict__ keys, const float* __restrict__ ws_attn,
                    float* __restrict__ ctx_out) {
    const int hc = blockIdx.x, b = blockIdx.y;
    const int tid = threadIdx.x;
    __shared__ float attn_s[T_];
    for (int t = tid; t < T_; t += 256) attn_s[t] = ws_attn[b * T_ + t];
    __syncthreads();
    const int hi = tid & 63, tq = tid >> 6;
    const int h = hc * 64 + hi;
    const float* kp = keys + (size_t)b * T_ * H_ + h;
    float acc = 0.0f;
    for (int t = tq; t < T_; t += 4)
        acc += attn_s[t] * kp[(size_t)t * H_];
    __shared__ float part[4][64];
    part[tq][hi] = acc;
    __syncthreads();
    if (tq == 0)
        ctx_out[b * H_ + h] = part[0][hi] + part[1][hi] + part[2][hi] + part[3][hi];
}

// ---------------------------------------------------------------------------
extern "C" void kernel_launch(void* const* d_in, const int* in_sizes, int n_in,
                              void* d_out, int out_size, void* d_ws, size_t ws_size,
                              hipStream_t stream) {
    const float* query = (const float*)d_in[0];   // (16,1,1024)
    const float* keys  = (const float*)d_in[1];   // (16,4096,1024)
    const float* Wq    = (const float*)d_in[2];   // (1024,1024)
    const float* Wk    = (const float*)d_in[3];   // (1024,1024)
    const float* v     = (const float*)d_in[4];   // (1024,)

    float* out      = (float*)d_out;
    float* ctx_out  = out;                 // 16*1024 floats
    float* attn_out = out + B_ * H_;       // 16*4096 floats

    char*   ws      = (char*)d_ws;
    bf16_t* wsB     = (bf16_t*)ws;                               // 2 MB (swizzled Wk)
    float*  qproj   = (float*)(ws + (size_t)2 * 1024 * 1024);    // 64 KB
    float*  score   = qproj + B_ * H_;                           // 256 KB
    float*  ws_attn = score + B_ * T_;                           // 256 KB

    // Allow 264 KB dynamic LDS (WGP supports 320 KB). Non-stream call: not
    // captured by graph capture; deterministic and idempotent.
    (void)hipFuncSetAttribute((const void*)score_kernel,
                              hipFuncAttributeMaxDynamicSharedMemorySize,
                              SMEM_BYTES);

    qproj_kernel     <<<dim3(B_),        dim3(256), 0, stream>>>(query, Wq, qproj);
    wk_swizzle_kernel<<<dim3(256),       dim3(256), 0, stream>>>(Wk, wsB);
    score_kernel     <<<dim3(T_ / ROWS_PER_WG, B_), dim3(256), SMEM_BYTES, stream>>>
                        (keys, wsB, qproj, v, score);
    softmax_kernel   <<<dim3(B_),        dim3(256), 0, stream>>>(score, attn_out, ws_attn);
    context_kernel   <<<dim3(16, B_),    dim3(256), 0, stream>>>(keys, ws_attn, ctx_out);
}
